// OmniAdaptiveFeature_83640193122753
// MI455X (gfx1250) — matrix operand order repaired and checked by reading, not compiled, not run on hardware
//
#include <hip/hip_runtime.h>
#include <hip/hip_bf16.h>
#include <math.h>

typedef __bf16 bf16_t;
typedef __bf16 v16bf __attribute__((ext_vector_type(16)));
typedef float  v8f   __attribute__((ext_vector_type(8)));

static constexpr int    kN    = 131072;
static constexpr int    kC    = 256;
static constexpr size_t kNC   = (size_t)kN * kC;
static constexpr float  kEps  = 1e-5f;

// ---------------------------------------------------------------------------
// Elementwise / utility kernels
// ---------------------------------------------------------------------------
__global__ void k_zero_f32(float* __restrict__ p, size_t n) {
  size_t i = (size_t)blockIdx.x * blockDim.x + threadIdx.x;
  size_t s = (size_t)gridDim.x * blockDim.x;
  for (; i < n; i += s) p[i] = 0.f;
}

__global__ void k_f32_to_bf16(const float* __restrict__ src, bf16_t* __restrict__ dst,
                              size_t n) {
  size_t i = (size_t)blockIdx.x * blockDim.x + threadIdx.x;
  size_t s = (size_t)gridDim.x * blockDim.x;
  for (; i < n; i += s) dst[i] = (bf16_t)src[i];
}

// ---------------------------------------------------------------------------
// GEMM: Out[N,256] (f32) = A[N,256] (bf16) @ W[256,256] (f32, converted to bf16)
// Block = 256 threads (8 wave32). Each wave: 16 rows x 64-col slab.
// Weight slab staged in LDS in WMMA B-fragment layout (32 KB).
//   B frag (K=32 x N=16, bf16): lane l -> col = l&15, khalf = l>>4;
//     dword v holds K = khalf*16 + 2v, 2v+1.
//   A frag (M=16 x K=32, bf16): lane l -> row = l&15, h = l>>4;
//     dword v holds K = h*16 + 2v (+ pair).
//   C tile (16x16 f32): lane l -> col = l&15, VGPR q -> row = (l>>4)*8 + q.
// BETA is a compile-time flag: 1 accumulates into Out (split concat-GEMM of the
// fuse layer) with all 8 read-back loads issued before a single wait.
// ---------------------------------------------------------------------------
template <int BETA>
__global__ __launch_bounds__(256)
void k_gemm_bf16(const bf16_t* __restrict__ A, const float* __restrict__ Wg,
                 float* __restrict__ Out) {
  __shared__ v16bf sW[4 * 8 * 32];  // [ntile][kslice][lane] -> 16 bf16 = 32 KB

  const int tid  = threadIdx.x;
  const int lane = tid & 31;
  const int wave = tid >> 5;
  const int h    = lane >> 4;
  const int nOff = blockIdx.y * 64;

  // Cooperative fill: convert f32 weights into bf16 B-fragments in LDS.
  bf16_t* sWp = (bf16_t*)sW;
  for (int idx = tid; idx < 4 * 8 * 32 * 8; idx += 256) {
    int v  = idx & 7;
    int l  = (idx >> 3) & 31;
    int ks = (idx >> 8) & 7;
    int nt = idx >> 11;
    int n  = nOff + nt * 16 + (l & 15);
    int k  = ks * 32 + (l >> 4) * 16 + v * 2;
    sWp[2 * idx + 0] = (bf16_t)Wg[(size_t)(k + 0) * kC + n];
    sWp[2 * idx + 1] = (bf16_t)Wg[(size_t)(k + 1) * kC + n];
  }
  __syncthreads();

  const int rowBase = blockIdx.x * 128 + wave * 16;

  // Load all A fragments for this wave's 16 rows across full K=256 (64 VGPRs).
  const bf16_t* arow = A + (size_t)(rowBase + (lane & 15)) * kC;
  v16bf afrag[8];
#pragma unroll
  for (int ks = 0; ks < 8; ++ks) {
#pragma unroll
    for (int v = 0; v < 8; ++v) {
      int k = ks * 32 + (h << 4) + (v << 1);
      afrag[ks][2 * v + 0] = arow[k + 0];
      afrag[ks][2 * v + 1] = arow[k + 1];
    }
  }

#pragma unroll
  for (int nt = 0; nt < 4; ++nt) {
    v8f acc = {};
#pragma unroll
    for (int ks = 0; ks < 8; ++ks) {
      v16bf b = sW[(nt * 8 + ks) * 32 + lane];
      acc = __builtin_amdgcn_wmma_f32_16x16x32_bf16(false, afrag[ks], false, b,
                                                    (short)0, acc, false, false);
    }
    int n = nOff + nt * 16 + (lane & 15);
    float* orow = Out + (size_t)(rowBase + h * 8) * kC + n;
    if (BETA) {
      // Issue all 8 read-back loads up front; single wait; no per-element branch.
      float old[8];
#pragma unroll
      for (int q = 0; q < 8; ++q) old[q] = orow[(size_t)q * kC];
#pragma unroll
      for (int q = 0; q < 8; ++q) orow[(size_t)q * kC] = acc[q] + old[q];
    } else {
#pragma unroll
      for (int q = 0; q < 8; ++q) orow[(size_t)q * kC] = acc[q];
    }
  }
}

// ---------------------------------------------------------------------------
// BatchNorm stats: per-column sum & sum-of-squares (biased variance later).
// One block = 256 rows, one thread per channel, two atomics (sum, sum-sq) per channel/block.
// ---------------------------------------------------------------------------
__global__ void k_col_stats(const float* __restrict__ Y, float* __restrict__ stats) {
  int c  = threadIdx.x;
  int r0 = blockIdx.x * 256;
  float s = 0.f, s2 = 0.f;
  for (int r = 0; r < 256; ++r) {
    float v = Y[(size_t)(r0 + r) * kC + c];
    s += v; s2 += v * v;
  }
  atomicAdd(&stats[c], s);
  atomicAdd(&stats[kC + c], s2);
}

// BN (training-mode, biased var) -> ReLU -> optional residual / f32 / bf16 out
__global__ void k_bn_relu(const float* __restrict__ Y, const float* __restrict__ stats,
                          const float* __restrict__ g, const float* __restrict__ b,
                          float* outf, bf16_t* outbf, const float* residual) {
  size_t i = (size_t)blockIdx.x * blockDim.x + threadIdx.x;
  size_t s = (size_t)gridDim.x * blockDim.x;
  const float invN = 1.f / (float)kN;
  for (; i < kNC; i += s) {
    int c = (int)(i & (kC - 1));
    float mean = stats[c] * invN;
    float var  = stats[kC + c] * invN - mean * mean;
    float v = (Y[i] - mean) * rsqrtf(var + kEps) * g[c] + b[c];
    v = fmaxf(v, 0.f);
    if (residual) v += residual[i];
    if (outf)  outf[i]  = v;
    if (outbf) outbf[i] = (bf16_t)v;
  }
}

// ---------------------------------------------------------------------------
// Segment (cluster) operations — atomics resolve mostly in the 192 MB L2
// ---------------------------------------------------------------------------
__global__ void k_seg_count(const int* __restrict__ cl, float* __restrict__ cnt) {
  int i = blockIdx.x * blockDim.x + threadIdx.x;
  if (i < kN) atomicAdd(&cnt[cl[i]], 1.f);
}

__global__ void k_seg_add(const float* __restrict__ src, const int* __restrict__ cl,
                          float* __restrict__ seg) {
  size_t i = (size_t)blockIdx.x * blockDim.x + threadIdx.x;
  size_t s = (size_t)gridDim.x * blockDim.x;
  for (; i < kNC; i += s) {
    int n  = (int)(i >> 8);
    int ch = (int)(i & 255);
    atomicAdd(&seg[(size_t)cl[n] * kC + ch], src[i]);
  }
}

// pw - scatter_mean(pw)[c]  -> bf16 (next GEMM operand)
__global__ void k_sub_mean_bf16(const float* __restrict__ pw, const float* __restrict__ seg,
                                const float* __restrict__ cnt, const int* __restrict__ cl,
                                bf16_t* __restrict__ out) {
  size_t i = (size_t)blockIdx.x * blockDim.x + threadIdx.x;
  size_t s = (size_t)gridDim.x * blockDim.x;
  for (; i < kNC; i += s) {
    int n  = (int)(i >> 8);
    int ch = (int)(i & 255);
    int c  = cl[n];
    float m = seg[(size_t)c * kC + ch] / fmaxf(cnt[c], 1.f);
    out[i] = (bf16_t)(pw[i] - m);
  }
}

__device__ inline void atomicMaxF(float* addr, float val) {
  int* ai = (int*)addr;
  int cur = *ai;
  while (__int_as_float(cur) < val) {
    int prev = atomicCAS(ai, cur, __float_as_int(val));
    if (prev == cur) break;
    cur = prev;
  }
}

__global__ void k_init_neginf(float* mx) { *mx = -3.4e38f; }

__global__ void k_max_reduce(const float* __restrict__ Y, float* __restrict__ mx) {
  __shared__ float sm[256];
  size_t i = (size_t)blockIdx.x * blockDim.x + threadIdx.x;
  size_t s = (size_t)gridDim.x * blockDim.x;
  float m = -3.4e38f;
  for (; i < kNC; i += s) m = fmaxf(m, Y[i]);
  sm[threadIdx.x] = m;
  __syncthreads();
  for (int t = 128; t > 0; t >>= 1) {
    if (threadIdx.x < t) sm[threadIdx.x] = fmaxf(sm[threadIdx.x], sm[threadIdx.x + t]);
    __syncthreads();
  }
  if (threadIdx.x == 0) atomicMaxF(mx, sm[0]);
}

__global__ void k_exp_inplace(float* __restrict__ Y, const float* __restrict__ mx) {
  size_t i = (size_t)blockIdx.x * blockDim.x + threadIdx.x;
  size_t s = (size_t)gridDim.x * blockDim.x;
  float m = *mx;
  for (; i < kNC; i += s) Y[i] = expf(Y[i] - m);
}

// w = exp / (segment_sum(exp)[c] + 1e-6)
__global__ void k_softmax_weights(float* __restrict__ Y, const float* __restrict__ seg,
                                  const int* __restrict__ cl) {
  size_t i = (size_t)blockIdx.x * blockDim.x + threadIdx.x;
  size_t s = (size_t)gridDim.x * blockDim.x;
  for (; i < kNC; i += s) {
    int n  = (int)(i >> 8);
    int ch = (int)(i & 255);
    Y[i] = Y[i] / (seg[(size_t)cl[n] * kC + ch] + 1e-6f);
  }
}

__global__ void k_mul_inplace(float* __restrict__ A, const float* __restrict__ B) {
  size_t i = (size_t)blockIdx.x * blockDim.x + threadIdx.x;
  size_t s = (size_t)gridDim.x * blockDim.x;
  for (; i < kNC; i += s) A[i] *= B[i];
}

// fused += adp[:, br] * seg[c[n]]   (gather-back + adaptive blend, no atomics)
__global__ void k_gather_acc(const float* __restrict__ seg, const int* __restrict__ cl,
                             const float* __restrict__ adp, int br,
                             float* __restrict__ fused) {
  size_t i = (size_t)blockIdx.x * blockDim.x + threadIdx.x;
  size_t s = (size_t)gridDim.x * blockDim.x;
  for (; i < kNC; i += s) {
    int n  = (int)(i >> 8);
    int ch = (int)(i & 255);
    fused[i] += adp[(size_t)n * 3 + br] * seg[(size_t)cl[n] * kC + ch];
  }
}

// adp = softmax(feat @ adp_w + adp_b) over 3 scales; one point per thread
__global__ void k_adp(const float* __restrict__ feat, const float* __restrict__ aw,
                      const float* __restrict__ ab, float* __restrict__ adp) {
  int n = blockIdx.x * blockDim.x + threadIdx.x;
  if (n >= kN) return;
  float a0 = ab[0], a1 = ab[1], a2 = ab[2];
  const float* f = feat + (size_t)n * kC;
  for (int ch = 0; ch < kC; ++ch) {
    float v = f[ch];
    a0 += v * aw[ch * 3 + 0];
    a1 += v * aw[ch * 3 + 1];
    a2 += v * aw[ch * 3 + 2];
  }
  float m  = fmaxf(a0, fmaxf(a1, a2));
  float e0 = expf(a0 - m), e1 = expf(a1 - m), e2 = expf(a2 - m);
  float inv = 1.f / (e0 + e1 + e2);
  adp[(size_t)n * 3 + 0] = e0 * inv;
  adp[(size_t)n * 3 + 1] = e1 * inv;
  adp[(size_t)n * 3 + 2] = e2 * inv;
}

// ---------------------------------------------------------------------------
extern "C" void kernel_launch(void* const* d_in, const int* in_sizes, int n_in,
                              void* d_out, int out_size, void* d_ws, size_t ws_size,
                              hipStream_t stream) {
  (void)in_sizes; (void)n_in; (void)out_size; (void)ws_size;

  const float* feat     = (const float*)d_in[0];
  const int*   clusters = (const int*)  d_in[1];
  const float* proj_w   = (const float*)d_in[2];
  const float* proj_g   = (const float*)d_in[3];
  const float* proj_b   = (const float*)d_in[4];
  const float* lw_w     = (const float*)d_in[5];
  const float* lw_g     = (const float*)d_in[6];
  const float* lw_b     = (const float*)d_in[7];
  const float* w_w      = (const float*)d_in[8];
  const float* adp_w    = (const float*)d_in[9];
  const float* adp_b    = (const float*)d_in[10];
  const float* fuse_w   = (const float*)d_in[11];
  const float* fuse_g   = (const float*)d_in[12];
  const float* fuse_b   = (const float*)d_in[13];
  float* out = (float*)d_out;

  // Workspace layout (~642 MB)
  char* p = (char*)d_ws;
  bf16_t* featbf = (bf16_t*)p; p += kNC * 2;   // feat in bf16 (reused as fused_bf at end)
  bf16_t* bufAbf = (bf16_t*)p; p += kNC * 2;   // bf16 staging for second GEMM operands
  float* Y     = (float*)p; p += kNC * 4;      // GEMM out / pw / softmax weights
  float* SEG   = (float*)p; p += kNC * 4;      // segment accumulator (L2-resident)
  float* PF    = (float*)p; p += kNC * 4;      // projected features
  float* FUSED = (float*)p; p += kNC * 4;      // adaptive-blended features
  float* cnt   = (float*)p; p += (size_t)kN * 4;
  float* adp   = (float*)p; p += (size_t)kN * 3 * 4;
  float* stats = (float*)p; p += 2 * kC * 4;
  float* mx    = (float*)p; p += 64;

  const int EW = 4096;                 // grid-stride elementwise blocks
  dim3 gGemm(kN / 128, kC / 64);       // 1024 x 4 blocks, 256 threads

  k_f32_to_bf16<<<EW, 256, 0, stream>>>(feat, featbf, kNC);
  k_zero_f32  <<<EW, 256, 0, stream>>>(FUSED, kNC);
  k_adp       <<<(kN + 255) / 256, 256, 0, stream>>>(feat, adp_w, adp_b, adp);

  for (int i = 0; i < 3; ++i) {
    const int* cl = clusters + (size_t)i * kN;

    // pw = relu(bn(feat @ lw_w[i]))
    k_gemm_bf16<0><<<gGemm, 256, 0, stream>>>(featbf, lw_w + (size_t)i * kC * kC, Y);
    k_zero_f32 <<<2, 256, 0, stream>>>(stats, 2 * kC);
    k_col_stats<<<kN / 256, 256, 0, stream>>>(Y, stats);
    k_bn_relu  <<<EW, 256, 0, stream>>>(Y, stats, lw_g + i * kC, lw_b + i * kC,
                                        Y, (bf16_t*)nullptr, (const float*)nullptr);
    // pw -= scatter_mean(pw)[c]   (as bf16 operand)
    k_zero_f32 <<<EW, 256, 0, stream>>>(cnt, kN);
    k_seg_count<<<kN / 256, 256, 0, stream>>>(cl, cnt);
    k_zero_f32 <<<EW, 256, 0, stream>>>(SEG, kNC);
    k_seg_add  <<<EW, 256, 0, stream>>>(Y, cl, SEG);
    k_sub_mean_bf16<<<EW, 256, 0, stream>>>(Y, SEG, cnt, cl, bufAbf);

    // pw = pw @ w_w[i]; segment softmax with global-max stabilization
    k_gemm_bf16<0><<<gGemm, 256, 0, stream>>>(bufAbf, w_w + (size_t)i * kC * kC, Y);
    k_init_neginf<<<1, 1, 0, stream>>>(mx);
    k_max_reduce <<<1024, 256, 0, stream>>>(Y, mx);
    k_exp_inplace<<<EW, 256, 0, stream>>>(Y, mx);
    k_zero_f32   <<<EW, 256, 0, stream>>>(SEG, kNC);
    k_seg_add    <<<EW, 256, 0, stream>>>(Y, cl, SEG);
    k_softmax_weights<<<EW, 256, 0, stream>>>(Y, SEG, cl);

    // pf = relu(bn(feat @ proj_w[i])) * w;  feats_i = gather(segment_sum(pf))
    k_gemm_bf16<0><<<gGemm, 256, 0, stream>>>(featbf, proj_w + (size_t)i * kC * kC, PF);
    k_zero_f32 <<<2, 256, 0, stream>>>(stats, 2 * kC);
    k_col_stats<<<kN / 256, 256, 0, stream>>>(PF, stats);
    k_bn_relu  <<<EW, 256, 0, stream>>>(PF, stats, proj_g + i * kC, proj_b + i * kC,
                                        PF, (bf16_t*)nullptr, (const float*)nullptr);
    k_mul_inplace<<<EW, 256, 0, stream>>>(PF, Y);
    k_zero_f32   <<<EW, 256, 0, stream>>>(SEG, kNC);
    k_seg_add    <<<EW, 256, 0, stream>>>(PF, cl, SEG);
    k_gather_acc <<<EW, 256, 0, stream>>>(SEG, cl, adp, i, FUSED);
  }

  // f = relu(bn(feat @ proj_w[3])) -> bf16
  k_gemm_bf16<0><<<gGemm, 256, 0, stream>>>(featbf, proj_w + (size_t)3 * kC * kC, Y);
  k_zero_f32 <<<2, 256, 0, stream>>>(stats, 2 * kC);
  k_col_stats<<<kN / 256, 256, 0, stream>>>(Y, stats);
  k_bn_relu  <<<EW, 256, 0, stream>>>(Y, stats, proj_g + 3 * kC, proj_b + 3 * kC,
                                      (float*)nullptr, bufAbf, (const float*)nullptr);
  // fused -> bf16 (featbf no longer needed: reuse)
  k_f32_to_bf16<<<EW, 256, 0, stream>>>(FUSED, featbf, kNC);

  // concat([f, fused]) @ fuse_w  ==  f @ fuse_w[:C] + fused @ fuse_w[C:]
  k_gemm_bf16<0><<<gGemm, 256, 0, stream>>>(bufAbf, fuse_w, Y);
  k_gemm_bf16<1><<<gGemm, 256, 0, stream>>>(featbf, fuse_w + (size_t)kC * kC, Y);

  // out = relu(bn(Y)) + feat
  k_zero_f32 <<<2, 256, 0, stream>>>(stats, 2 * kC);
  k_col_stats<<<kN / 256, 256, 0, stream>>>(Y, stats);
  k_bn_relu  <<<EW, 256, 0, stream>>>(Y, stats, fuse_g, fuse_b, out,
                                      (bf16_t*)nullptr, feat);
}
